// encoder_unit_83554293776725
// MI455X (gfx1250) — matrix operand order repaired and checked by reading, not compiled, not run on hardware
//
#include <hip/hip_runtime.h>

// ---- problem dims (fixed by reference) ----
#define SDIM 2048
#define EDIM 1024
#define HHEADS 16
#define DHEAD 64
#define FDIM 4096   // 4*E

typedef __bf16 bf16;
typedef __attribute__((ext_vector_type(8)))  __bf16 v8bf;
typedef __attribute__((ext_vector_type(16))) __bf16 v16bf;
typedef __attribute__((ext_vector_type(8)))  float  v8f;
typedef __attribute__((ext_vector_type(4)))  unsigned int v4u;
typedef __attribute__((ext_vector_type(8)))  int v8i;
typedef __attribute__((ext_vector_type(4)))  int v4i;

#if defined(__has_builtin)
#if __has_builtin(__builtin_amdgcn_tensor_load_to_lds) && __has_builtin(__builtin_amdgcn_s_wait_tensorcnt)
#define HAVE_TDM 1
#endif
#endif
#ifndef HAVE_TDM
#define HAVE_TDM 0
#endif

union FragBF { v16bf v; v8bf h[2]; };

__device__ __forceinline__ v8f wmma_bf16(v16bf a, v16bf b, v8f c) {
  // D = A(16x32 bf16) * B(32x16 bf16) + C(16x16 f32)
  return __builtin_amdgcn_wmma_f32_16x16x32_bf16(false, a, false, b, (short)0, c, false, false);
}

#if HAVE_TDM
// ---- Tensor DMA descriptor helpers (CDNA5 ISA ch. 8.3-8.6) ----
// group0: count=1 (valid), lds_addr, global byte address (57b), type=2 ("image")
__device__ __forceinline__ v4u tdm_g0(unsigned ldsOff, unsigned long long ga) {
  return (v4u){1u, ldsOff, (unsigned)ga,
               (unsigned)((ga >> 32) & 0x01FFFFFFull) | 0x80000000u};
}
// group1: data_size=1 (2 bytes), huge tensor dims (no OOB clip), 2D tile
// tile_dim0 = k-contiguous width, tile_dim1 = rows, dim0 row stride in elements
__device__ __forceinline__ v8i tdm_g1(unsigned tile0, unsigned tile1,
                                      unsigned long long strideElems) {
  const unsigned long long td0 = 0x100000ull, td1 = 0x100000ull;
  unsigned w0 = 0x00010000u;                                              // data_size=1
  unsigned w1 = (unsigned)((td0 & 0xFFFFull) << 16);                      // tensor_dim0 lo
  unsigned w2 = (unsigned)(((td0 >> 16) & 0xFFFFull) | ((td1 & 0xFFFFull) << 16));
  unsigned w3 = (unsigned)(((td1 >> 16) & 0xFFFFull) | ((tile0 & 0xFFFFu) << 16));
  unsigned w4 = tile1 & 0xFFFFu;                                          // tile_dim1, tile_dim2=0
  unsigned w5 = (unsigned)(strideElems & 0xFFFFFFFFull);                  // dim0 stride lo
  unsigned w6 = (unsigned)((strideElems >> 32) & 0xFFFFull);              // stride hi, dim1 stride=0
  return (v8i){(int)w0, (int)w1, (int)w2, (int)w3, (int)w4, (int)w5, (int)w6, 0};
}
__device__ __forceinline__ v4i tdm_g2(void) { return (v4i){0x100000, 0, 0, 0}; }
__device__ __forceinline__ v4i tdm_g3(void) { return (v4i){0, 0, 0, 0}; }
__device__ __forceinline__ v8i tdm_zero8(void) { return (v8i){0, 0, 0, 0, 0, 0, 0, 0}; }
__device__ __forceinline__ unsigned lds_off(const void* p) {
  return (unsigned)(unsigned long long)(uintptr_t)p;
}
__device__ __forceinline__ void tdm_load(unsigned ldsOff, unsigned long long ga,
                                         unsigned tile0, unsigned tile1,
                                         unsigned long long strideElems) {
  // clang-23 / therock-10.0 arity: (g0 v4u, g1 v8i, g2 v4i, g3 v4i, v8i, i32 cpol)
  __builtin_amdgcn_tensor_load_to_lds(tdm_g0(ldsOff, ga), tdm_g1(tile0, tile1, strideElems),
                                      tdm_g2(), tdm_g3(), tdm_zero8(), 0);
}
#endif

// ---- DPP 16-lane butterfly reductions (no LDS traffic, pure VALU) ----
template <int CTRL>
__device__ __forceinline__ float dpp_xf(float v) {
  return __builtin_bit_cast(
      float, __builtin_amdgcn_update_dpp(0, __builtin_bit_cast(int, v), CTRL, 0xF, 0xF, true));
}
__device__ __forceinline__ float rmax16(float v) {
  v = fmaxf(v, dpp_xf<0xB1>(v));   // quad_perm [1,0,3,2]
  v = fmaxf(v, dpp_xf<0x4E>(v));   // quad_perm [2,3,0,1]
  v = fmaxf(v, dpp_xf<0x141>(v));  // row_half_mirror
  v = fmaxf(v, dpp_xf<0x140>(v));  // row_mirror
  return v;
}
__device__ __forceinline__ float rsum16(float v) {
  v += dpp_xf<0xB1>(v);
  v += dpp_xf<0x4E>(v);
  v += dpp_xf<0x141>(v);
  v += dpp_xf<0x140>(v);
  return v;
}

// ---------------- elementwise helpers ----------------
__global__ __launch_bounds__(256) void add_emb_kernel(
    const float* __restrict__ xw, const float* __restrict__ xp,
    float* __restrict__ embF, bf16* __restrict__ embB) {
  int i = blockIdx.x * 256 + threadIdx.x;
  float v = xw[i] + xp[i];
  embF[i] = v;
  embB[i] = (bf16)v;
}

__global__ __launch_bounds__(256) void f2bf_kernel(
    const float* __restrict__ in, bf16* __restrict__ out, int n) {
  int i = blockIdx.x * 256 + threadIdx.x;
  if (i < n) out[i] = (bf16)in[i];
}

// convert + transpose per-head projection weights: in [H][E][D] -> out [H][D][E]
// so every GEMM B-matrix is [N][K] row-major (TDM-stageable without transpose)
__global__ __launch_bounds__(256) void f2bf_tr_kernel(
    const float* __restrict__ in, bf16* __restrict__ out) {
  int hd = blockIdx.x;
  int h = hd / DHEAD, d = hd % DHEAD;
  const float* src = in + (long)h * EDIM * DHEAD + d;
  bf16* dst = out + (long)h * DHEAD * EDIM + (long)d * EDIM;
#pragma unroll
  for (int i = 0; i < 4; ++i) {
    int e = i * 256 + threadIdx.x;
    dst[e] = (bf16)src[(long)e * DHEAD];
  }
}

// ---------------- generic bf16 WMMA GEMM ----------------
// C[m,n] = sum_k A[m,k]*B[n*ldb + k] + bias[n]  (B is [N,K] row-major).
// Block: 256 threads (8 waves), tile 128(M) x 64(N), K-step 64, wave tile 32x32.
// A and B tiles staged by the Tensor Data Mover when available.
template <bool OUT_BF16, bool RELU>
__global__ __launch_bounds__(256) void gemm_bf16_wmma(
    const bf16* __restrict__ A, int lda,
    const bf16* __restrict__ B, long ldb, long bHeadStride,
    const float* __restrict__ bias, long biasHeadStride,
    void* __restrict__ Cv, int ldc, long cHeadStride, int K) {
  __shared__ __align__(16) bf16 sA[128 * 64];  // [m][k]
  __shared__ __align__(16) bf16 sB[64 * 64];   // [n][k]

  const int h = blockIdx.z;
  const bf16* Bh = B + (long)h * bHeadStride;
  const float* bh = bias + (long)h * biasHeadStride;
  const int m0 = blockIdx.x * 128, n0 = blockIdx.y * 64;
  const int tid = threadIdx.x;
  const int lane = tid & 31, wv = tid >> 5;
  const int half = lane >> 4, lm = lane & 15;
  const int wm = wv >> 1, wn = wv & 1;

  v8f acc[2][2] = {};

#if HAVE_TDM
  const unsigned long long baseA =
      (unsigned long long)(uintptr_t)(A + (long)m0 * lda);
  const unsigned long long baseB =
      (unsigned long long)(uintptr_t)(Bh + (long)n0 * ldb);
  const unsigned ldsA = lds_off(&sA[0]);
  const unsigned ldsB = lds_off(&sB[0]);
#else
  const int arow = tid >> 1, akb = (tid & 1) * 32;
  const int brow = tid >> 2, bkb = (tid & 3) * 16;
#endif

  for (int k0 = 0; k0 < K; k0 += 64) {
    __syncthreads();
#if HAVE_TDM
    if (tid < 32) {  // wave 0 drives the Tensor Data Mover
      tdm_load(ldsA, baseA + (unsigned long long)k0 * 2, 64u, 128u, (unsigned long long)lda);
      tdm_load(ldsB, baseB + (unsigned long long)k0 * 2, 64u, 64u, (unsigned long long)ldb);
      __builtin_amdgcn_s_wait_tensorcnt(0);
    }
#else
    {
      const v8bf* pa = (const v8bf*)(A + (long)(m0 + arow) * lda + k0 + akb);
#pragma unroll
      for (int i = 0; i < 4; ++i) *(v8bf*)&sA[arow * 64 + akb + i * 8] = pa[i];
      const v8bf* pb = (const v8bf*)(Bh + (long)(n0 + brow) * ldb + k0 + bkb);
#pragma unroll
      for (int i = 0; i < 2; ++i) *(v8bf*)&sB[brow * 64 + bkb + i * 8] = pb[i];
    }
#endif
    __syncthreads();

#pragma unroll
    for (int kc = 0; kc < 2; ++kc) {
      FragBF fa[2], fb[2];
#pragma unroll
      for (int mt = 0; mt < 2; ++mt) {
        int row = wm * 32 + mt * 16 + lm;
        fa[mt].h[0] = *(const v8bf*)&sA[row * 64 + kc * 32 + 8 * half];       // K=8*half+e
        fa[mt].h[1] = *(const v8bf*)&sA[row * 64 + kc * 32 + 16 + 8 * half];  // K=16+8*half+e
      }
#pragma unroll
      for (int nt = 0; nt < 2; ++nt) {
        int n = wn * 32 + nt * 16 + lm;
        fb[nt].h[0] = *(const v8bf*)&sB[n * 64 + kc * 32 + 8 * half];         // K=8*half+e
        fb[nt].h[1] = *(const v8bf*)&sB[n * 64 + kc * 32 + 16 + 8 * half];
      }
#pragma unroll
      for (int mt = 0; mt < 2; ++mt)
#pragma unroll
        for (int nt = 0; nt < 2; ++nt)
          acc[mt][nt] = wmma_bf16(fa[mt].v, fb[nt].v, acc[mt][nt]);
    }
  }

  // epilogue: C-matrix layout -> row = 8*half + j, col = lm
  bf16*  Cb = (bf16*)Cv  + (long)h * cHeadStride;
  float* Cf = (float*)Cv + (long)h * cHeadStride;
#pragma unroll
  for (int mt = 0; mt < 2; ++mt)
#pragma unroll
    for (int nt = 0; nt < 2; ++nt)
#pragma unroll
      for (int jv = 0; jv < 8; ++jv) {
        long row = m0 + wm * 32 + mt * 16 + 8 * half + jv;
        int  col = n0 + wn * 32 + nt * 16 + lm;
        float vv = acc[mt][nt][jv] + bh[col];
        if (RELU) vv = fmaxf(vv, 0.f);
        if (OUT_BF16) Cb[row * (long)ldc + col] = (bf16)vv;
        else          Cf[row * (long)ldc + col] = vv;
      }
}

// ---------------- flash attention ----------------
// Block = 4 waves = 4 query tiles (64 q rows) of one head; K/V kv-chunk staging
// is shared across the 4 waves (K row-major for score B-frags, V transposed with
// packed b32 stores for the P@V B-frags). Per-wave sP re-stripes softmax output
// from accumulator layout into A-fragment layout.
__global__ __launch_bounds__(128) void attn_flash_kernel(
    const bf16* __restrict__ Q, const bf16* __restrict__ K,
    const bf16* __restrict__ V, bf16* __restrict__ Z) {
  __shared__ __align__(16) bf16 sK[32 * 64];      // [kv][d] row-major
  __shared__ __align__(16) bf16 sVt[64 * 32];     // [d][kv]
  __shared__ __align__(16) bf16 sP[4][16 * 32];   // per-wave P tile [m][kv]

  const int h = blockIdx.y;
  const int tid = threadIdx.x, wv = tid >> 5, lane = tid & 31;
  const int half = lane >> 4, lm = lane & 15;
  const int r = blockIdx.x * 4 + wv;              // 16-row query tile index
  const long base = (long)h * SDIM * DHEAD;
  const int srow = tid >> 2, sc0 = (tid & 3) * 16;   // K staging: 32 rows x 4 chunks
  const int vrow = (tid >> 3) * 2, vc0 = (tid & 7) * 8;  // V staging: kv-row pairs x 8 d's

  // Q fragments (two 32-wide K-chunks of D=64), kept in registers
  FragBF aq[2];
#pragma unroll
  for (int c = 0; c < 2; ++c) {
    const bf16* qp = Q + base + (long)(r * 16 + lm) * DHEAD + c * 32 + 8 * half;
    aq[c].h[0] = *(const v8bf*)qp;
    aq[c].h[1] = *(const v8bf*)(qp + 16);
  }

  float mrow[8], lrow[8];
#pragma unroll
  for (int j = 0; j < 8; ++j) { mrow[j] = -1e30f; lrow[j] = 0.f; }
  v8f o[4] = {};

  for (int j0 = 0; j0 < SDIM; j0 += 32) {
    __syncthreads();  // all waves done reading previous sK/sVt
    // cooperative staging
    {
      const v8bf* pk = (const v8bf*)(K + base + (long)(j0 + srow) * DHEAD + sc0);
      v8bf ka = pk[0], kb = pk[1];
      *(v8bf*)&sK[srow * 64 + sc0]     = ka;
      *(v8bf*)&sK[srow * 64 + sc0 + 8] = kb;
      v8bf va = *(const v8bf*)(V + base + (long)(j0 + vrow) * DHEAD + vc0);
      v8bf vb = *(const v8bf*)(V + base + (long)(j0 + vrow + 1) * DHEAD + vc0);
#pragma unroll
      for (int i = 0; i < 8; ++i) {   // packed transpose: two kv per b32 store
        __bf16 x0 = va[i], x1 = vb[i];
        unsigned u = (unsigned)__builtin_bit_cast(unsigned short, x0) |
                     ((unsigned)__builtin_bit_cast(unsigned short, x1) << 16);
        *(unsigned*)&sVt[(vc0 + i) * 32 + vrow] = u;
      }
    }
    __syncthreads();

    // scores: two 16x16 tiles (kv rows j0..j0+31), K-dim = D via 2 WMMAs each
    v8f st[2] = {};
#pragma unroll
    for (int jn = 0; jn < 2; ++jn)
#pragma unroll
      for (int c = 0; c < 2; ++c) {
        FragBF fb;
        const bf16* kp = &sK[(jn * 16 + lm) * 64 + c * 32 + 8 * half];
        fb.h[0] = *(const v8bf*)kp;
        fb.h[1] = *(const v8bf*)(kp + 16);
        st[jn] = wmma_bf16(aq[c].v, fb.v, st[jn]);
      }
    // online softmax (acc rows = 8*half + jv; cols striped across 16-lane half)
    float scale[8];
#pragma unroll
    for (int jv = 0; jv < 8; ++jv) {
      float rm = rmax16(fmaxf(st[0][jv], st[1][jv]));
      float mnew = fmaxf(mrow[jv], rm);
      scale[jv] = __expf(mrow[jv] - mnew);
      float e0 = __expf(st[0][jv] - mnew);
      float e1 = __expf(st[1][jv] - mnew);
      lrow[jv] = lrow[jv] * scale[jv] + rsum16(e0 + e1);
      mrow[jv] = mnew;
      sP[wv][(8 * half + jv) * 32 + lm]      = (bf16)e0;
      sP[wv][(8 * half + jv) * 32 + 16 + lm] = (bf16)e1;
    }
#pragma unroll
    for (int nt = 0; nt < 4; ++nt)
#pragma unroll
      for (int jv = 0; jv < 8; ++jv) o[nt][jv] = o[nt][jv] * scale[jv];
    // O += P(16x32) @ V(32x64); sP is wave-private (in-order LDS within wave)
    FragBF fp;
    fp.h[0] = *(const v8bf*)&sP[wv][lm * 32 + 8 * half];
    fp.h[1] = *(const v8bf*)&sP[wv][lm * 32 + 16 + 8 * half];
#pragma unroll
    for (int nt = 0; nt < 4; ++nt) {
      FragBF fv;
      int n = nt * 16 + lm;
      fv.h[0] = *(const v8bf*)&sVt[n * 32 + 8 * half];
      fv.h[1] = *(const v8bf*)&sVt[n * 32 + 16 + 8 * half];
      o[nt] = wmma_bf16(fp.v, fv.v, o[nt]);
    }
  }

  // finalize: divide by row sum, apply reference's post-softmax 1/sqrt(D)
  float inv[8];
#pragma unroll
  for (int jv = 0; jv < 8; ++jv) inv[jv] = 0.125f / lrow[jv];
#pragma unroll
  for (int nt = 0; nt < 4; ++nt)
#pragma unroll
    for (int jv = 0; jv < 8; ++jv) {
      long row = r * 16 + 8 * half + jv;
      Z[row * (long)EDIM + h * DHEAD + nt * 16 + lm] = (bf16)(o[nt][jv] * inv[jv]);
    }
}

// ---------------- fused residual + LayerNorm ----------------
__global__ __launch_bounds__(256) void ln_residual_kernel(
    const float* __restrict__ X, const float* __restrict__ Yres,
    const float* __restrict__ g, const float* __restrict__ be,
    float* __restrict__ outF, bf16* __restrict__ outB) {
  __shared__ float s1[256], s2[256];
  const int row = blockIdx.x, t = threadIdx.x;
  float vals[4], a = 0.f, b = 0.f;
#pragma unroll
  for (int i = 0; i < 4; ++i) {
    int c = i * 256 + t;
    float v = X[(long)row * EDIM + c] + Yres[(long)row * EDIM + c];
    vals[i] = v; a += v; b += v * v;
  }
  s1[t] = a; s2[t] = b;
  __syncthreads();
  for (int off = 128; off > 0; off >>= 1) {
    if (t < off) { s1[t] += s1[t + off]; s2[t] += s2[t + off]; }
    __syncthreads();
  }
  float mean = s1[0] * (1.f / EDIM);
  float var  = s2[0] * (1.f / EDIM) - mean * mean;
  float rstd = rsqrtf(var + 1e-5f);
#pragma unroll
  for (int i = 0; i < 4; ++i) {
    int c = i * 256 + t;
    float ov = (vals[i] - mean) * rstd * g[c] + be[c];
    outF[(long)row * EDIM + c] = ov;
    if (outB) outB[(long)row * EDIM + c] = (bf16)ov;
  }
}

// ---------------- host-side orchestration ----------------
extern "C" void kernel_launch(void* const* d_in, const int* in_sizes, int n_in,
                              void* d_out, int out_size, void* d_ws, size_t ws_size,
                              hipStream_t stream) {
  (void)in_sizes; (void)n_in; (void)out_size; (void)ws_size;
  const float* xw  = (const float*)d_in[0];
  const float* xp  = (const float*)d_in[1];
  const float* Wq  = (const float*)d_in[2];
  const float* bq  = (const float*)d_in[3];
  const float* Wk  = (const float*)d_in[4];
  const float* bk  = (const float*)d_in[5];
  const float* Wv  = (const float*)d_in[6];
  const float* bv  = (const float*)d_in[7];
  const float* W0  = (const float*)d_in[8];
  const float* b0  = (const float*)d_in[9];
  const float* g1  = (const float*)d_in[10];
  const float* be1 = (const float*)d_in[11];
  const float* W1  = (const float*)d_in[12];
  const float* b1  = (const float*)d_in[13];
  const float* W2  = (const float*)d_in[14];
  const float* b2  = (const float*)d_in[15];
  const float* g2  = (const float*)d_in[16];
  const float* be2 = (const float*)d_in[17];

  char* p = (char*)d_ws;
  auto take = [&](size_t bytes) -> char* {
    char* r = p; p += (bytes + 255) & ~(size_t)255; return r;
  };
  float* embF  = (float*)take((size_t)SDIM * EDIM * 4);
  bf16*  embB  = (bf16*) take((size_t)SDIM * EDIM * 2);
  bf16*  WqB   = (bf16*) take((size_t)HHEADS * EDIM * DHEAD * 2);  // [H][D][E]
  bf16*  WkB   = (bf16*) take((size_t)HHEADS * EDIM * DHEAD * 2);  // [H][D][E]
  bf16*  WvB   = (bf16*) take((size_t)HHEADS * EDIM * DHEAD * 2);  // [H][D][E]
  bf16*  W0B   = (bf16*) take((size_t)EDIM * EDIM * 2);
  bf16*  W1B   = (bf16*) take((size_t)FDIM * EDIM * 2);
  bf16*  W2B   = (bf16*) take((size_t)EDIM * FDIM * 2);
  bf16*  QB    = (bf16*) take((size_t)HHEADS * SDIM * DHEAD * 2);
  bf16*  KB    = (bf16*) take((size_t)HHEADS * SDIM * DHEAD * 2);
  bf16*  VB    = (bf16*) take((size_t)HHEADS * SDIM * DHEAD * 2);
  bf16*  ZB    = (bf16*) take((size_t)SDIM * EDIM * 2);
  float* z0F   = (float*)take((size_t)SDIM * EDIM * 4);
  float* out1F = (float*)take((size_t)SDIM * EDIM * 4);
  bf16*  out1B = (bf16*) take((size_t)SDIM * EDIM * 2);
  bf16*  h1B   = (bf16*) take((size_t)SDIM * FDIM * 2);
  float* out2F = (float*)take((size_t)SDIM * EDIM * 4);

  // 1) embeddings + weight conversion to bf16 (QKV weights also transposed to [H][D][E])
  add_emb_kernel<<<SDIM * EDIM / 256, 256, 0, stream>>>(xw, xp, embF, embB);
  f2bf_tr_kernel<<<HHEADS * DHEAD, 256, 0, stream>>>(Wq, WqB);
  f2bf_tr_kernel<<<HHEADS * DHEAD, 256, 0, stream>>>(Wk, WkB);
  f2bf_tr_kernel<<<HHEADS * DHEAD, 256, 0, stream>>>(Wv, WvB);
  f2bf_kernel<<<(EDIM * EDIM) / 256, 256, 0, stream>>>(W0, W0B, EDIM * EDIM);
  f2bf_kernel<<<(FDIM * EDIM) / 256, 256, 0, stream>>>(W1, W1B, FDIM * EDIM);
  f2bf_kernel<<<(EDIM * FDIM) / 256, 256, 0, stream>>>(W2, W2B, EDIM * FDIM);

  // 2) per-head QKV projections: [S,E] @ WqT[h] ([D,E] = [N,K] row-major)
  dim3 gqkv(SDIM / 128, DHEAD / 64, HHEADS);
  gemm_bf16_wmma<true, false><<<gqkv, 256, 0, stream>>>(
      embB, EDIM, WqB, EDIM, (long)DHEAD * EDIM, bq, DHEAD,
      QB, DHEAD, (long)SDIM * DHEAD, EDIM);
  gemm_bf16_wmma<true, false><<<gqkv, 256, 0, stream>>>(
      embB, EDIM, WkB, EDIM, (long)DHEAD * EDIM, bk, DHEAD,
      KB, DHEAD, (long)SDIM * DHEAD, EDIM);
  gemm_bf16_wmma<true, false><<<gqkv, 256, 0, stream>>>(
      embB, EDIM, WvB, EDIM, (long)DHEAD * EDIM, bv, DHEAD,
      VB, DHEAD, (long)SDIM * DHEAD, EDIM);

  // 3) flash attention -> Z (head-concatenated [S,E], bf16); 4 q-tiles per block
  attn_flash_kernel<<<dim3(SDIM / 64, HHEADS), 128, 0, stream>>>(QB, KB, VB, ZB);

  // 4) output projection: Z @ W0.T + b0  (W0 is [N,K] row-major already)
  gemm_bf16_wmma<false, false><<<dim3(SDIM / 128, EDIM / 64, 1), 256, 0, stream>>>(
      ZB, EDIM, W0B, EDIM, 0, b0, 0, z0F, EDIM, 0, EDIM);

  // 5) out1 = LN(z0 + emb)
  ln_residual_kernel<<<SDIM, 256, 0, stream>>>(z0F, embF, g1, be1, out1F, out1B);

  // 6) FFN1: relu(out1 @ W1.T + b1) -> bf16 hidden
  gemm_bf16_wmma<true, true><<<dim3(SDIM / 128, FDIM / 64, 1), 256, 0, stream>>>(
      out1B, EDIM, W1B, EDIM, 0, b1, 0, h1B, FDIM, 0, EDIM);

  // 7) FFN2: hidden @ W2.T + b2 -> fp32
  gemm_bf16_wmma<false, false><<<dim3(SDIM / 128, EDIM / 64, 1), 256, 0, stream>>>(
      h1B, FDIM, W2B, FDIM, 0, b2, 0, out2F, EDIM, 0, FDIM);

  // 8) final LN(out1 + out2) -> d_out (fp32)
  ln_residual_kernel<<<SDIM, 256, 0, stream>>>(out1F, out2F, g2, be2, (float*)d_out, nullptr);
}
